// Decoder_32615981645904
// MI455X (gfx1250) — matrix operand order repaired and checked by reading, not compiled, not run on hardware
//
#include <hip/hip_runtime.h>
#include <hip/hip_bf16.h>
#include <math.h>

// ---------------------------------------------------------------------------
// Show-Attend-Tell decoder on MI455X (gfx1250, wave32).
// All GEMMs: bf16 WMMA (v_wmma_f32_16x16x32_bf16), fp32 accumulate.
// Each wave computes a 32x(16*NSUB) C macro-tile: 2 A-frags x NSUB B-frags.
// The column count NSUB is a TEMPLATE parameter (hand loop-unswitching) so
// the K-loop body is branch-free: no PHI copies, accumulators stay pinned,
// EXEC stays all-1s for WMMA, and B loads stay global (not flat).
// Weights packed once per call into WMMA B-fragment lane order.
// ---------------------------------------------------------------------------

typedef __bf16 bf16_t;
typedef __bf16 v16bf __attribute__((ext_vector_type(16)));
typedef float  v8f   __attribute__((ext_vector_type(8)));
typedef float  v4f   __attribute__((ext_vector_type(4)));

#define BB   64      // batch
#define LL   196     // attention locations
#define TT   20      // caption length
#define NS   19      // scan steps = T-1
#define EE   512     // embed dim
#define DD   512     // decoder dim
#define AAD  512     // attention dim
#define NC   512     // encoder dim
#define VV   10000   // vocab

// wave id within grid; threadIdx.x>>5 is wave-uniform on wave32, so
// readfirstlane is exact and moves the value (and all derivees) to SGPRs.
__device__ inline int wave_id_scalar()
{
    return __builtin_amdgcn_readfirstlane((int)(blockIdx.x * 8 + (threadIdx.x >> 5)));
}

// ---------------------------------------------------------------------------
// Pack fp32 weight W[k, n] (row-major, Ksrc x N) into bf16 WMMA B-fragment
// order for a K-dim of Ktot (supports vertical concat via koff).
// Fragment layout (ISA 7.12.2, 16-bit B 32x16, wave32):
//   lane = 16*((k_in_block/8)%2) + (n%16),  element e = 8*(k_in_block/16) + k_in_block%8
// Packed index: ((tile_n*(Ktot/32) + kblock)*32 + lane)*16 + e
// ---------------------------------------------------------------------------
__global__ void pack_w_kernel(const float* __restrict__ W, bf16_t* __restrict__ Wp,
                              int Ksrc, int N, int Ktot, int koff)
{
    int idx = blockIdx.x * blockDim.x + threadIdx.x;
    if (idx >= Ksrc * N) return;
    int k = idx / N, n = idx - k * N;
    int kk = k + koff;
    int tile_n = n >> 4;
    int kb = kk >> 5;
    int kr = kk & 31;
    int lanep = (((kr >> 3) & 1) << 4) + (n & 15);
    int e = ((kr >> 4) << 3) + (kr & 7);
    size_t dst = (((size_t)tile_n * (size_t)(Ktot >> 5) + kb) * 32 + lanep) * 16 + e;
    Wp[dst] = (bf16_t)W[(size_t)idx];
}

// ---------------------------------------------------------------------------
// Load one 16x32 A-fragment (fp32 -> bf16 in-register).
// Element e -> k = 16*(e>=8) + 8*(lane>=16) + (e&7); ap already includes
// row*K + half*8.
// ---------------------------------------------------------------------------
__device__ __forceinline__ v16bf load_a_frag(const float* __restrict__ ap)
{
    v4f f0 = *(const v4f*)(ap);
    v4f f1 = *(const v4f*)(ap + 4);
    v4f f2 = *(const v4f*)(ap + 16);
    v4f f3 = *(const v4f*)(ap + 20);
    v16bf a;
    #pragma unroll
    for (int j = 0; j < 4; ++j) {
        a[j]      = (bf16_t)f0[j];
        a[4 + j]  = (bf16_t)f1[j];
        a[8 + j]  = (bf16_t)f2[j];
        a[12 + j] = (bf16_t)f3[j];
    }
    return a;
}

// ---------------------------------------------------------------------------
// Branch-free GEMM core for a 32x(16*NSUB) macro-tile.
// ---------------------------------------------------------------------------
template <int NSUB>
__device__ __forceinline__ void gemm_core(
    const float* __restrict__ A, const bf16_t* __restrict__ Wp,
    const float* __restrict__ bias, float* __restrict__ C,
    int K, int ldc, int act, int gm, int tn0, int lane)
{
    const int l15 = lane & 15;
    const int half = lane >> 4;
    const int KB = K >> 5;

    v8f acc0[NSUB], acc1[NSUB];
    #pragma unroll
    for (int j = 0; j < NSUB; ++j) {
        float bv = bias ? bias[((tn0 + j) << 4) + l15] : 0.0f;
        #pragma unroll
        for (int e = 0; e < 8; ++e) { acc0[j][e] = bv; acc1[j][e] = bv; }
    }

    const float* pa0 = A + ((size_t)(gm << 5) + l15) * K + half * 8;
    const float* pa1 = pa0 + (size_t)16 * K;
    const bf16_t* pw[NSUB];
    #pragma unroll
    for (int j = 0; j < NSUB; ++j) {
        pw[j] = Wp + ((size_t)(tn0 + j) * KB * 32 + lane) * 16;
        __builtin_prefetch(pw[j], 0, 1);       // global_prefetch_b8 (prologue)
    }

    #pragma clang loop unroll(disable)
    for (int kb = 0; kb < KB; ++kb) {
        v16bf a0 = load_a_frag(pa0);
        v16bf a1 = load_a_frag(pa1);
        pa0 += 32; pa1 += 32;
        #pragma unroll
        for (int j = 0; j < NSUB; ++j) {
            v16bf b = *(const v16bf*)pw[j];
            pw[j] += 512;                      // 32 lanes * 16 elems
            acc0[j] = __builtin_amdgcn_wmma_f32_16x16x32_bf16(
                          false, a0, false, b, (short)0, acc0[j], false, false);
            acc1[j] = __builtin_amdgcn_wmma_f32_16x16x32_bf16(
                          false, a1, false, b, (short)0, acc1[j], false, false);
        }
    }

    // C-fragment (f32 16x16): element e -> row = 8*half + e, col = lane&15
    const int mb0 = (gm << 5) + half * 8;
    #pragma unroll
    for (int j = 0; j < NSUB; ++j) {
        const int cb = ((tn0 + j) << 4) + l15;
        #pragma unroll
        for (int e = 0; e < 8; ++e) {
            float v0 = acc0[j][e];
            float v1 = acc1[j][e];
            if (act == 1)      { v0 = tanhf(v0); v1 = tanhf(v1); }
            else if (act == 2) { v0 = 1.0f / (1.0f + expf(-v0));
                                 v1 = 1.0f / (1.0f + expf(-v1)); }
            C[(size_t)(mb0 + e) * ldc + cb]      = v0;
            C[(size_t)(mb0 + 16 + e) * ldc + cb] = v1;
        }
    }
}

// ---------------------------------------------------------------------------
// WMMA GEMM: C[M,N] = act( A[M,K] (fp32) * Wp (packed bf16) + bias )
// M mult of 32, N mult of 16, K mult of 32.
// act: 0=none, 1=tanh, 2=sigmoid.  C row stride = ldc.
// ---------------------------------------------------------------------------
__global__ __launch_bounds__(256) void wmma_gemm_kernel(
    const float* __restrict__ A, const bf16_t* __restrict__ Wp,
    const float* __restrict__ bias, float* __restrict__ C,
    int M, int K, int N, int ldc, int act)
{
    const int lane = threadIdx.x & 31;
    const int wid  = wave_id_scalar();    // SGPR
    const int NT   = N >> 4;              // 16-col tiles
    const int NG   = (NT + 3) >> 2;       // 64-col groups
    const int MG   = M >> 5;              // 32-row groups
    if (wid >= MG * NG) return;           // scalar branch
    const int gm = wid / NG, gn = wid - gm * NG;     // SGPRs
    const int tn0 = gn << 2;
    int nsub = NT - tn0; if (nsub > 4) nsub = 4;     // SGPR

    switch (nsub) {                        // scalar dispatch, outside hot loop
    case 4:  gemm_core<4>(A, Wp, bias, C, K, ldc, act, gm, tn0, lane); break;
    case 3:  gemm_core<3>(A, Wp, bias, C, K, ldc, act, gm, tn0, lane); break;
    case 2:  gemm_core<2>(A, Wp, bias, C, K, ldc, act, gm, tn0, lane); break;
    default: gemm_core<1>(A, Wp, bias, C, K, ldc, act, gm, tn0, lane); break;
    }
}

// --------------------------- small VALU kernels ----------------------------

__global__ void embed_gather_kernel(const float* __restrict__ emb,
                                    const int* __restrict__ cap,
                                    float* __restrict__ out)   // (B,T,E)
{
    int idx = blockIdx.x * blockDim.x + threadIdx.x;
    if (idx >= BB * TT * EE) return;
    int e = idx & (EE - 1);
    int bt = idx >> 9;
    out[idx] = emb[(size_t)cap[bt] * EE + e];
}

__global__ void mean_feat_kernel(const float* __restrict__ feat,
                                 float* __restrict__ mf)       // (B,NC)
{
    int idx = blockIdx.x * blockDim.x + threadIdx.x;
    if (idx >= BB * NC) return;
    int ch = idx & (NC - 1), b = idx >> 9;
    const float* p = feat + (size_t)b * LL * NC + ch;
    float s = 0.0f;
    for (int l = 0; l < LL; ++l) s += p[(size_t)l * NC];
    mf[idx] = s * (1.0f / (float)LL);
}

__global__ void combine_bias_kernel(const float* __restrict__ bih,
                                    const float* __restrict__ bhh,
                                    const float* __restrict__ blh,
                                    const float* __restrict__ blz,
                                    float* __restrict__ bg, float* __restrict__ bo)
{
    int i = blockIdx.x * blockDim.x + threadIdx.x;
    if (i < 4 * DD) bg[i] = bih[i] + bhh[i];
    if (i < EE)     bo[i] = blh[i] + blz[i];
}

// e[b,l] = relu(feat_proj[b,l,:] + hproj[b,:]) . Wa_full + ba_full
// one wave32 per (b,l), lane-parallel dot + xor-shuffle reduction
__global__ __launch_bounds__(256) void attn_score_kernel(
    const float* __restrict__ fproj, const float* __restrict__ hproj,
    const float* __restrict__ wfull, const float* __restrict__ bfull,
    float* __restrict__ esc)
{
    int wid = wave_id_scalar();           // SGPR
    int lane = threadIdx.x & 31;
    if (wid >= BB * LL) return;           // scalar branch
    int b = wid / LL;
    const float* fp = fproj + (size_t)wid * AAD;
    const float* hp = hproj + (size_t)b * AAD;
    float acc = 0.0f;
    for (int a = lane; a < AAD; a += 32) {
        float v = fp[a] + hp[a];
        v = v > 0.0f ? v : 0.0f;
        acc += v * wfull[a];
    }
    #pragma unroll
    for (int off = 16; off > 0; off >>= 1) acc += __shfl_xor(acc, off, 32);
    if (lane == 0) esc[wid] = acc + bfull[0];
}

// per-b softmax over L, alpha -> d_out, z[b,:] = sum_l alpha * features[b,l,:]
__global__ __launch_bounds__(256) void softmax_z_kernel(
    const float* __restrict__ esc, const float* __restrict__ feat,
    float* __restrict__ z, float* __restrict__ alpha_out)
{
    __shared__ float red[256];
    __shared__ float sal[LL];
    int b = blockIdx.x, tid = threadIdx.x;
    float v = (tid < LL) ? esc[b * LL + tid] : -3.0e38f;
    red[tid] = v; __syncthreads();
    for (int s = 128; s > 0; s >>= 1) {
        if (tid < s) red[tid] = fmaxf(red[tid], red[tid + s]);
        __syncthreads();
    }
    float mx = red[0]; __syncthreads();
    float ex = (tid < LL) ? expf(v - mx) : 0.0f;
    red[tid] = ex; __syncthreads();
    for (int s = 128; s > 0; s >>= 1) {
        if (tid < s) red[tid] += red[tid + s];
        __syncthreads();
    }
    float inv = 1.0f / red[0];
    if (tid < LL) {
        float al = ex * inv;
        sal[tid] = al;
        alpha_out[(size_t)b * (NS * LL) + tid] = al;   // (b, t, l) with t-base folded in
    }
    __syncthreads();
    for (int ch = tid; ch < NC; ch += 256) {
        float acc = 0.0f;
        const float* fp = feat + (size_t)b * LL * NC + ch;
        for (int l = 0; l < LL; ++l) acc += sal[l] * fp[(size_t)l * NC];
        z[b * NC + ch] = acc;
    }
}

// z *= beta; build lstm_in = [x_t, z, h] and outcat z-half
__global__ void build_inputs_kernel(const float* __restrict__ beta,
                                    const float* __restrict__ embeds,
                                    const float* __restrict__ h,
                                    float* __restrict__ z,
                                    float* __restrict__ lstm_in,
                                    float* __restrict__ outcat, int t)
{
    int idx = blockIdx.x * blockDim.x + threadIdx.x;
    if (idx >= BB * DD) return;
    int b = idx >> 9, d = idx & 511;
    float zg = z[idx] * beta[idx];
    z[idx] = zg;
    lstm_in[(size_t)b * 1536 + d]            = embeds[((size_t)b * TT + t) * EE + d];
    lstm_in[(size_t)b * 1536 + EE + d]       = zg;
    lstm_in[(size_t)b * 1536 + EE + NC + d]  = h[idx];
    outcat[(size_t)b * 1024 + DD + d]        = zg;
}

// LSTM cell (gate order i,f,g,o); writes h,c in place and outcat h-half
__global__ void lstm_update_kernel(const float* __restrict__ gates,
                                   float* __restrict__ h, float* __restrict__ c,
                                   float* __restrict__ outcat)
{
    int idx = blockIdx.x * blockDim.x + threadIdx.x;
    if (idx >= BB * DD) return;
    int b = idx >> 9, d = idx & 511;
    const float* g = gates + (size_t)b * 4 * DD;
    float gi = g[d], gf = g[DD + d], gg = g[2 * DD + d], go = g[3 * DD + d];
    float si = 1.0f / (1.0f + expf(-gi));
    float sf = 1.0f / (1.0f + expf(-gf));
    float so = 1.0f / (1.0f + expf(-go));
    float cn = sf * c[idx] + si * tanhf(gg);
    float hn = so * tanhf(cn);
    c[idx] = cn;
    h[idx] = hn;
    outcat[(size_t)b * 1024 + d] = hn;
}

__global__ void add_x_kernel(float* __restrict__ oute,
                             const float* __restrict__ embeds, int t)
{
    int idx = blockIdx.x * blockDim.x + threadIdx.x;
    if (idx >= BB * EE) return;
    int b = idx >> 9, d = idx & 511;
    oute[idx] += embeds[((size_t)b * TT + t) * EE + d];
}

// ---------------------------------------------------------------------------

extern "C" void kernel_launch(void* const* d_in, const int* in_sizes, int n_in,
                              void* d_out, int out_size, void* d_ws, size_t ws_size,
                              hipStream_t stream)
{
    (void)in_sizes; (void)n_in; (void)out_size; (void)ws_size;

    const float* features = (const float*)d_in[0];
    const int*   captions = (const int*)d_in[1];
    const float* emb_tab  = (const float*)d_in[2];
    const float* Wa_enc   = (const float*)d_in[3];
    const float* ba_enc   = (const float*)d_in[4];
    const float* Wa_dec   = (const float*)d_in[5];
    const float* ba_dec   = (const float*)d_in[6];
    const float* Wa_full  = (const float*)d_in[7];
    const float* ba_full  = (const float*)d_in[8];
    const float* W_beta   = (const float*)d_in[9];
    const float* b_beta   = (const float*)d_in[10];
    const float* W_ih     = (const float*)d_in[11];
    const float* b_ih     = (const float*)d_in[12];
    const float* W_hh     = (const float*)d_in[13];
    const float* b_hh     = (const float*)d_in[14];
    const float* W_init_h = (const float*)d_in[15];
    const float* b_init_h = (const float*)d_in[16];
    const float* W_init_c = (const float*)d_in[17];
    const float* b_init_c = (const float*)d_in[18];
    const float* W_Lo     = (const float*)d_in[19];
    const float* b_Lo     = (const float*)d_in[20];
    const float* W_Lh     = (const float*)d_in[21];
    const float* b_Lh     = (const float*)d_in[22];
    const float* W_Lz     = (const float*)d_in[23];
    const float* b_Lz     = (const float*)d_in[24];

    float* out_logits = (float*)d_out;                          // (B, NS, V)
    float* out_alphas = out_logits + (size_t)BB * NS * VV;      // (B, NS, L)

    // -------- workspace carve-up (256B aligned) --------
    char* ws = (char*)d_ws;
    size_t off = 0;
    auto carve = [&](size_t bytes) -> void* {
        void* p = ws + off;
        off = (off + bytes + 255) & ~(size_t)255;
        return p;
    };
    bf16_t* Waenc_p = (bf16_t*)carve((size_t)512 * 512 * 2);
    bf16_t* Wadec_p = (bf16_t*)carve((size_t)512 * 512 * 2);
    bf16_t* Wbeta_p = (bf16_t*)carve((size_t)512 * 512 * 2);
    bf16_t* Winh_p  = (bf16_t*)carve((size_t)512 * 512 * 2);
    bf16_t* Winc_p  = (bf16_t*)carve((size_t)512 * 512 * 2);
    bf16_t* Wcat_p  = (bf16_t*)carve((size_t)1536 * 2048 * 2); // [W_ih; W_hh]
    bf16_t* Woc_p   = (bf16_t*)carve((size_t)1024 * 512 * 2);  // [W_Lh; W_Lz]
    bf16_t* WLo_p   = (bf16_t*)carve((size_t)512 * VV * 2);
    float*  bg      = (float*)carve((size_t)4 * DD * 4);       // b_ih + b_hh
    float*  bo      = (float*)carve((size_t)EE * 4);           // b_Lh + b_Lz
    float*  embeds  = (float*)carve((size_t)BB * TT * EE * 4);
    float*  fproj   = (float*)carve((size_t)BB * LL * AAD * 4);
    float*  meanf   = (float*)carve((size_t)BB * NC * 4);
    float*  h       = (float*)carve((size_t)BB * DD * 4);
    float*  c       = (float*)carve((size_t)BB * DD * 4);
    float*  hproj   = (float*)carve((size_t)BB * AAD * 4);
    float*  esc     = (float*)carve((size_t)BB * LL * 4);
    float*  z       = (float*)carve((size_t)BB * NC * 4);
    float*  beta    = (float*)carve((size_t)BB * NC * 4);
    float*  lstm_in = (float*)carve((size_t)BB * 1536 * 4);
    float*  gates   = (float*)carve((size_t)BB * 4 * DD * 4);
    float*  outcat  = (float*)carve((size_t)BB * 1024 * 4);
    float*  oute    = (float*)carve((size_t)BB * EE * 4);

    auto pack = [&](const float* W, bf16_t* Wp, int Ksrc, int N, int Ktot, int koff) {
        int n = Ksrc * N;
        pack_w_kernel<<<(n + 255) / 256, 256, 0, stream>>>(W, Wp, Ksrc, N, Ktot, koff);
    };
    auto gemm = [&](const float* A, const bf16_t* Wp, const float* bias, float* C,
                    int M, int K, int N, int ldc, int act) {
        int NG = ((N >> 4) + 3) >> 2;          // 64-col groups
        int tiles = (M >> 5) * NG;             // 32-row groups x col groups
        wmma_gemm_kernel<<<(tiles + 7) / 8, 256, 0, stream>>>(A, Wp, bias, C, M, K, N, ldc, act);
    };

    // -------- one-time prep --------
    pack(Wa_enc,   Waenc_p, 512,  512,   512,  0);
    pack(Wa_dec,   Wadec_p, 512,  512,   512,  0);
    pack(W_beta,   Wbeta_p, 512,  512,   512,  0);
    pack(W_init_h, Winh_p,  512,  512,   512,  0);
    pack(W_init_c, Winc_p,  512,  512,   512,  0);
    pack(W_ih,     Wcat_p,  1024, 2048,  1536, 0);
    pack(W_hh,     Wcat_p,  512,  2048,  1536, 1024);
    pack(W_Lh,     Woc_p,   512,  512,   1024, 0);
    pack(W_Lz,     Woc_p,   512,  512,   1024, 512);
    pack(W_Lo,     WLo_p,   512,  VV,    512,  0);

    combine_bias_kernel<<<(4 * DD + 255) / 256, 256, 0, stream>>>(b_ih, b_hh, b_Lh, b_Lz, bg, bo);
    embed_gather_kernel<<<(BB * TT * EE + 255) / 256, 256, 0, stream>>>(emb_tab, captions, embeds);
    mean_feat_kernel<<<(BB * NC + 255) / 256, 256, 0, stream>>>(features, meanf);

    gemm(meanf, Winh_p, b_init_h, h, BB, NC, DD, DD, 1);   // h0 = tanh(...)
    gemm(meanf, Winc_p, b_init_c, c, BB, NC, DD, DD, 1);   // c0 = tanh(...)
    gemm(features, Waenc_p, ba_enc, fproj, BB * LL, NC, AAD, AAD, 0);  // feat_proj

    // -------- 19 sequential decode steps --------
    for (int t = 0; t < NS; ++t) {
        gemm(h, Wadec_p, ba_dec, hproj, BB, DD, AAD, AAD, 0);
        attn_score_kernel<<<(BB * LL + 7) / 8, 256, 0, stream>>>(fproj, hproj, Wa_full, ba_full, esc);
        softmax_z_kernel<<<BB, 256, 0, stream>>>(esc, features, z, out_alphas + (size_t)t * LL);
        gemm(h, Wbeta_p, b_beta, beta, BB, DD, NC, NC, 2);  // sigmoid gate
        build_inputs_kernel<<<(BB * DD + 255) / 256, 256, 0, stream>>>(
            beta, embeds, h, z, lstm_in, outcat, t);
        gemm(lstm_in, Wcat_p, bg, gates, BB, 1536, 4 * DD, 4 * DD, 0);
        lstm_update_kernel<<<(BB * DD + 255) / 256, 256, 0, stream>>>(gates, h, c, outcat);
        gemm(outcat, Woc_p, bo, oute, BB, 1024, EE, EE, 0);
        add_x_kernel<<<(BB * EE + 255) / 256, 256, 0, stream>>>(oute, embeds, t);
        gemm(oute, WLo_p, b_Lo, out_logits + (size_t)t * VV, BB, EE, VV, NS * VV, 0);
    }
}
